// RobustHybridEEGModelV2_55920474194479
// MI455X (gfx1250) — matrix operand order repaired and checked by reading, not compiled, not run on hardware
//
#include <hip/hip_runtime.h>
#include <hip/hip_bf16.h>
#include <cstddef>
#include <cstdint>

// ---------------- constants ----------------
#define B_    128
#define C_    19
#define T_    1000
#define F_    16
#define NODES (B_ * C_)   // 2432
#define H_    8
#define D_    64
#define HD_   512

typedef __attribute__((ext_vector_type(16))) _Float16 v16h;
typedef __attribute__((ext_vector_type(8)))  float    v8f;

// ---------------- math helpers ----------------
__device__ __forceinline__ float elu_f(float x)  { return x > 0.f ? x : expm1f(x); }
__device__ __forceinline__ float gelu_f(float x) { return 0.5f * x * (1.f + erff(x * 0.70710678118654752f)); }
__device__ __forceinline__ float sigm_f(float x) { return 1.f / (1.f + expf(-x)); }
__device__ __forceinline__ float lrelu_f(float x){ return x > 0.f ? x : 0.2f * x; }

// ============================================================================
// Kernel A: fused EEGNet frontend.  One block per batch element.
// conv1(1x25)+bn1 and depthwise(C)+bn2 fused by factoring the per-filter
// affine bn1 through the channel sum.  Then ELU, pool4 (pre-SE; SE scale
// commutes with pooling), SE, separable conv + bn3 + ELU + temporal mean.
// ============================================================================
__global__ __launch_bounds__(256) void fe_kernel(
    const float* __restrict__ x,
    const float* __restrict__ w1,  const float* __restrict__ g1, const float* __restrict__ b1,
    const float* __restrict__ dw,  const float* __restrict__ g2, const float* __restrict__ b2,
    const float* __restrict__ sew1, const float* __restrict__ sew2,
    const float* __restrict__ sepw, const float* __restrict__ g3, const float* __restrict__ b3,
    float* __restrict__ yv)
{
  __shared__ _Float16 s_xc[F_][T_];     // 32 KB channel-mixed signal
  __shared__ float    s_p [F_][250];    // 16 KB pooled (pre-SE) activations
  __shared__ float    s_dw[F_][C_];
  __shared__ float    s_w1[F_][25];
  __shared__ float    s_sdw[F_], s_s[F_], s_u[4], s_s2[F_], s_acc[F_];

  const int b = blockIdx.x, tid = threadIdx.x;

  for (int i = tid; i < F_ * C_; i += blockDim.x) s_dw[i / C_][i % C_] = dw[i];
  for (int i = tid; i < F_ * 25; i += blockDim.x) s_w1[i / 25][i % 25] = w1[i];
  __syncthreads();
  if (tid < F_) {
    float sd = 0.f;
    for (int c = 0; c < C_; ++c) sd += s_dw[tid][c];
    s_sdw[tid] = sd; s_acc[tid] = 0.f;
  }
  const float* xb = x + (size_t)b * C_ * T_;
  for (int it = tid; it < F_ * T_; it += blockDim.x) {
    int f = it / T_, t = it % T_;
    float v = 0.f;
    for (int c = 0; c < C_; ++c) v += s_dw[f][c] * xb[c * T_ + t];
    s_xc[f][t] = (_Float16)v;
  }
  __syncthreads();
  for (int it = tid; it < F_ * 250; it += blockDim.x) {
    int f = it / 250, tt = it % 250;
    const float g1f = g1[f], cst = b1[f] * s_sdw[f], g2f = g2[f], b2f = b2[f];
    float acc = 0.f;
    for (int q = 0; q < 4; ++q) {
      int t = tt * 4 + q;
      float v = 0.f;
      int klo = max(0, 12 - t), khi = min(25, T_ + 12 - t);
      for (int k = klo; k < khi; ++k) v += s_w1[f][k] * (float)s_xc[f][t + k - 12];
      float y = g2f * (g1f * v + cst) + b2f;
      acc += elu_f(y);
    }
    s_p[f][tt] = acc;
  }
  __syncthreads();
  if (tid < F_) {
    float sum = 0.f;
    for (int tt = 0; tt < 250; ++tt) sum += s_p[tid][tt];
    s_s[tid] = sum * (1.0f / T_);
  }
  __syncthreads();
  if (tid < 4) {
    float u = 0.f;
    for (int f = 0; f < F_; ++f) u += s_s[f] * sew1[f * 4 + tid];
    s_u[tid] = gelu_f(u);
  }
  __syncthreads();
  if (tid < F_) {
    float v = 0.f;
    for (int j = 0; j < 4; ++j) v += s_u[j] * sew2[j * F_ + tid];
    s_s2[tid] = sigm_f(v);
  }
  __syncthreads();
  for (int it = tid; it < F_ * 250; it += blockDim.x) {
    int f = it / 250; s_p[f][it % 250] *= s_s2[f] * 0.25f;
  }
  __syncthreads();
  for (int it = tid; it < F_ * 251; it += blockDim.x) {
    int fo = it / 251, tp = it % 251;
    float o = 0.f;
    int klo = max(0, 8 - tp), khi = min(16, 258 - tp);
    for (int fi = 0; fi < F_; ++fi) {
      const float* wrow = sepw + fo * 256 + fi * 16;
      const float* prow = s_p[fi];
      for (int k = klo; k < khi; ++k) o += prow[tp - 8 + k] * wrow[k];
    }
    float y = g3[fo] * o + b3[fo];
    atomicAdd(&s_acc[fo], elu_f(y));
  }
  __syncthreads();
  if (tid < F_) yv[b * F_ + tid] = s_acc[tid] * (1.0f / 251.0f);
}

// ============================================================================
// Kernel B: per-node multiscale conv1d (15/25/35 taps) + bn + ELU, /3, mean_t.
// ============================================================================
__global__ __launch_bounds__(256) void ms_kernel(
    const float* __restrict__ x,
    const float* __restrict__ w0, const float* __restrict__ ga0, const float* __restrict__ bb0,
    const float* __restrict__ w1, const float* __restrict__ ga1, const float* __restrict__ bb1,
    const float* __restrict__ w2, const float* __restrict__ ga2, const float* __restrict__ bb2,
    float* __restrict__ feat)
{
  __shared__ float s_x[T_];
  __shared__ float s_w[F_ * 75];
  __shared__ float s_acc[F_];
  const int n = blockIdx.x, tid = threadIdx.x;
  const float* xr = x + (size_t)n * T_;
  for (int i = tid; i < T_; i += blockDim.x) s_x[i] = xr[i];
  for (int i = tid; i < F_ * 15; i += blockDim.x) s_w[i] = w0[i];
  for (int i = tid; i < F_ * 25; i += blockDim.x) s_w[F_ * 15 + i] = w1[i];
  for (int i = tid; i < F_ * 35; i += blockDim.x) s_w[F_ * 40 + i] = w2[i];
  if (tid < F_) s_acc[tid] = 0.f;
  __syncthreads();
  for (int f = 0; f < F_; ++f) {
    const float* wf0 = s_w + f * 15;
    const float* wf1 = s_w + F_ * 15 + f * 25;
    const float* wf2 = s_w + F_ * 40 + f * 35;
    const float ga0f = ga0[f], bb0f = bb0[f], ga1f = ga1[f], bb1f = bb1[f],
                ga2f = ga2[f], bb2f = bb2[f];
    float part = 0.f;
    for (int t = tid; t < T_; t += blockDim.x) {
      float v0 = 0.f, v1 = 0.f, v2 = 0.f;
      { int klo = max(0, 7 - t),  khi = min(15, T_ + 7 - t);
        for (int k = klo; k < khi; ++k) v0 += s_x[t - 7 + k]  * wf0[k]; }
      { int klo = max(0, 12 - t), khi = min(25, T_ + 12 - t);
        for (int k = klo; k < khi; ++k) v1 += s_x[t - 12 + k] * wf1[k]; }
      { int klo = max(0, 17 - t), khi = min(35, T_ + 17 - t);
        for (int k = klo; k < khi; ++k) v2 += s_x[t - 17 + k] * wf2[k]; }
      part += elu_f(ga0f * v0 + bb0f) + elu_f(ga1f * v1 + bb1f) + elu_f(ga2f * v2 + bb2f);
    }
    atomicAdd(&s_acc[f], part);
  }
  __syncthreads();
  if (tid < F_) feat[(size_t)n * 32 + 16 + tid] = s_acc[tid] * (1.0f / 3000.0f);
}

// ============================================================================
// Kernel C: broadcast yv into feat[:,0:16]; edge scores = sigmoid(feat@adj_w+b)
// ============================================================================
__global__ void feat_kernel(const float* __restrict__ yv, const float* __restrict__ adj_w,
                            const float* __restrict__ adj_b,
                            float* __restrict__ feat, float* __restrict__ scores)
{
  int n = blockIdx.x * blockDim.x + threadIdx.x;
  if (n >= NODES) return;
  int b = n / C_;
  float acc = adj_b[0];
  for (int q = 0; q < 16; ++q) {
    float v = yv[b * 16 + q];
    feat[(size_t)n * 32 + q] = v;
    acc += v * adj_w[q];
  }
  for (int q = 16; q < 32; ++q) acc += feat[(size_t)n * 32 + q] * adj_w[q];
  scores[n] = sigm_f(acc);
}

// ============================================================================
// WMMA GEMM with LDS-staged tiles: C[M,N] = A[M,K] * B[K,N], K in {32,64}.
// Block = 8 waves; one 16-row M-tile shared by the block, 8 16-col N-tiles
// (128-wide B panel).  The B panel is staged global->LDS with CDNA5
// GLOBAL_LOAD_ASYNC_TO_LDS_B128 (ASYNCcnt, s_wait_asynccnt) -- each lane
// supplies a 16B-aligned LDS byte offset (low 32 bits of the generic shared
// pointer) and a 64-bit global address.  A-tile staged with normal loads.
// Fragments are then built from LDS (ds loads) using the CDNA5 layouts:
//   A 16x32 f16: lane L -> M = L&15; elem e -> K = (e<8?e:e+8) + 8*(L>>4)
//   B 32x16 f16: lane L -> N = L&15; elem e -> K = (e<8?e:e+8) + 8*(L>>4)
//   D 16x16 f32: vgpr r -> M = r + 8*(L>>4); N = L&15
// ============================================================================
template <int K>
__global__ __launch_bounds__(256) void wmma_gemm(
    const float* __restrict__ A, const float* __restrict__ Bm,
    float* __restrict__ Cout, int M, int N)
{
  __shared__ __align__(16) float sA[16][K];      // <=4 KB
  __shared__ __align__(16) float sB[K][128];     // <=32 KB

  const int tid  = threadIdx.x;
  const int lane = tid & 31;
  const int wave = tid >> 5;                     // 0..7 -> local N tile
  const int ntiles  = N >> 4;
  const int mtiles  = M >> 4;
  const int ngroups = (ntiles + 7) >> 3;
  const int mt  = blockIdx.x / ngroups;
  const int grp = blockIdx.x % ngroups;
  const int n0  = grp * 128;                     // panel base column
  const int TW  = min(128, N - n0);              // panel width (multiple of 16)
  if (mt >= mtiles) return;

  // ---- stage B panel: async global->LDS, 16 bytes per lane per issue ----
  {
    const int tw4 = TW >> 2;                     // 4-float chunks per row
    for (int it = tid; it < K * tw4; it += 256) {
      int k  = it / tw4;
      int nc = (it % tw4) << 2;
      unsigned lds_off = (unsigned)(uintptr_t)&sB[k][nc];
      unsigned long long ga =
          (unsigned long long)(uintptr_t)(Bm + (size_t)k * N + n0 + nc);
      asm volatile("global_load_async_to_lds_b128 %0, %1, off"
                   :: "v"(lds_off), "v"(ga) : "memory");
    }
  }
  // ---- stage A tile with normal loads ----
  for (int it = tid; it < 16 * K; it += 256) {
    int m = it / K, k = it % K;
    sA[m][k] = A[(size_t)(mt * 16 + m) * K + k];
  }
  asm volatile("s_wait_asynccnt 0x0" ::: "memory");
  __syncthreads();

  // ---- compute: one 16x16 tile per wave ----
  const int hi = lane >> 4, l15 = lane & 15;
  const int ncol0 = n0 + wave * 16;
  if (ncol0 < N) {
    v8f acc = {};
#pragma unroll
    for (int k0 = 0; k0 < K; k0 += 32) {
      v16h a = {}, bfr = {};
#pragma unroll
      for (int e = 0; e < 16; ++e) {
        int ka = ((e < 8) ? e : (e + 8)) + hi * 8;
        a[e] = (_Float16)sA[l15][k0 + ka];
      }
#pragma unroll
      for (int e = 0; e < 16; ++e)
        bfr[e] = (_Float16)sB[k0 + ((e < 8) ? e : (e + 8)) + hi * 8][wave * 16 + l15];
      acc = __builtin_amdgcn_wmma_f32_16x16x32_f16(false, a, false, bfr,
                                                   (short)0, acc, false, false);
    }
    float* crow = Cout + (size_t)(mt * 16 + hi * 8) * N + ncol0 + l15;
#pragma unroll
    for (int r = 0; r < 8; ++r) crow[(size_t)r * N] = acc[r];
  }
}

// ============================================================================
// Kernel E: GAT attention, one fully-connected 19-node graph per block.
// ============================================================================
__global__ __launch_bounds__(256) void gat_kernel(
    const float* __restrict__ hbuf,
    const float* __restrict__ a_src, const float* __restrict__ a_dst,
    const float* __restrict__ scores, const float* __restrict__ bias,
    float* __restrict__ outbuf)
{
  __shared__ float s_sc[C_];
  __shared__ float s_as[C_][H_], s_ad[C_][H_];
  __shared__ float s_coef[C_][C_];
  __shared__ float s_hv[C_][D_];
  __shared__ float s_out[C_][D_];
  const int b = blockIdx.x, n0 = b * C_, tid = threadIdx.x;

  if (tid < C_) s_sc[tid] = scores[n0 + tid];
  for (int it = tid; it < C_ * H_ * 2; it += blockDim.x) {
    int which = it & 1, i = (it >> 1) / H_, hd = (it >> 1) % H_;
    const float* hv = hbuf + (size_t)(n0 + i) * HD_ + hd * D_;
    const float* av = (which ? a_dst : a_src) + hd * D_;
    float d = 0.f;
    for (int q = 0; q < D_; ++q) d += hv[q] * av[q];
    if (which) s_ad[i][hd] = d; else s_as[i][hd] = d;
  }
  for (int it = tid; it < C_ * D_; it += blockDim.x) s_out[it / D_][it % D_] = 0.f;
  __syncthreads();

  for (int hd = 0; hd < H_; ++hd) {
    if (tid < C_) {
      int j = tid;
      float ev[C_], m = -1e30f;
      for (int i = 0; i < C_; ++i) {
        float e = lrelu_f(s_as[i][hd] + s_ad[j][hd]);
        ev[i] = e; m = fmaxf(m, e);
      }
      float ssum = 0.f;
      for (int i = 0; i < C_; ++i) { ev[i] = expf(ev[i] - m); ssum += ev[i]; }
      float inv = 1.f / (ssum + 1e-16f);
      for (int i = 0; i < C_; ++i) s_coef[i][j] = ev[i] * inv * (s_sc[i] * s_sc[j]);
    }
    for (int it = tid; it < C_ * D_; it += blockDim.x)
      s_hv[it / D_][it % D_] = hbuf[(size_t)(n0 + it / D_) * HD_ + hd * D_ + (it % D_)];
    __syncthreads();
    for (int it = tid; it < C_ * D_; it += blockDim.x) {
      int j = it / D_, d = it % D_;
      float a = 0.f;
      for (int i = 0; i < C_; ++i) a += s_coef[i][j] * s_hv[i][d];
      s_out[j][d] += a;
    }
    __syncthreads();
  }
  for (int it = tid; it < C_ * D_; it += blockDim.x) {
    int j = it / D_, d = it % D_;
    outbuf[(size_t)(n0 + j) * D_ + d] = s_out[j][d] * (1.0f / H_) + bias[d];
  }
}

// ---------------- elementwise / pooling tails ----------------
__global__ void post1_kernel(const float* __restrict__ gat1out, const float* __restrict__ skipm,
                             const float* __restrict__ g, const float* __restrict__ bb,
                             const float* __restrict__ sb, float* __restrict__ out)
{
  int i = blockIdx.x * blockDim.x + threadIdx.x;
  if (i >= NODES * D_) return;
  int d = i & 63;
  out[i] = gelu_f(gat1out[i] * g[d] + bb[d] + skipm[i] + sb[d]);
}

__global__ void post2_kernel(const float* __restrict__ h2, const float* __restrict__ g,
                             const float* __restrict__ bb, float* __restrict__ out)
{
  int i = blockIdx.x * blockDim.x + threadIdx.x;
  if (i >= NODES * D_) return;
  int d = i & 63;
  out[i] = gelu_f(h2[i] * g[d] + bb[d] + h2[i]);
}

__global__ void pool_kernel(const float* __restrict__ hf, float* __restrict__ g)
{
  int i = blockIdx.x * blockDim.x + threadIdx.x;
  if (i >= B_ * D_) return;
  int b = i >> 6, d = i & 63;
  float s = 0.f;
  for (int j = 0; j < C_; ++j) s += hf[(size_t)(b * C_ + j) * D_ + d];
  g[i] = s * (1.0f / C_);
}

__global__ void cls_kernel(const float* __restrict__ act, const float* __restrict__ cb1,
                           const float* __restrict__ w2, const float* __restrict__ cb2,
                           float* __restrict__ out)
{
  int tid = threadIdx.x;
  if (tid >= B_ * 2) return;
  int b = tid >> 1, o = tid & 1;
  float s = cb2[o];
  for (int d = 0; d < D_; ++d) s += gelu_f(act[b * D_ + d] + cb1[d]) * w2[d * 2 + o];
  out[b * 2 + o] = s;
}

// ============================================================================
// launcher
// ============================================================================
extern "C" void kernel_launch(void* const* d_in, const int* in_sizes, int n_in,
                              void* d_out, int out_size, void* d_ws, size_t ws_size,
                              hipStream_t stream)
{
  (void)in_sizes; (void)n_in; (void)out_size; (void)ws_size;
  const float* x        = (const float*)d_in[0];
  const float* conv1_w  = (const float*)d_in[1];
  const float* bn1_g    = (const float*)d_in[2];
  const float* bn1_b    = (const float*)d_in[3];
  const float* depth_w  = (const float*)d_in[4];
  const float* bn2_g    = (const float*)d_in[5];
  const float* bn2_b    = (const float*)d_in[6];
  const float* se_w1    = (const float*)d_in[7];
  const float* se_w2    = (const float*)d_in[8];
  const float* sep_w    = (const float*)d_in[9];
  const float* bn3_g    = (const float*)d_in[10];
  const float* bn3_b    = (const float*)d_in[11];
  const float* ms_w0    = (const float*)d_in[12];
  const float* ms_g0    = (const float*)d_in[13];
  const float* ms_b0    = (const float*)d_in[14];
  const float* ms_w1    = (const float*)d_in[15];
  const float* ms_g1    = (const float*)d_in[16];
  const float* ms_b1    = (const float*)d_in[17];
  const float* ms_w2    = (const float*)d_in[18];
  const float* ms_g2    = (const float*)d_in[19];
  const float* ms_b2    = (const float*)d_in[20];
  const float* adj_w    = (const float*)d_in[21];
  const float* adj_b    = (const float*)d_in[22];
  const float* gat1_w   = (const float*)d_in[23];
  const float* gat1_as  = (const float*)d_in[24];
  const float* gat1_ad  = (const float*)d_in[25];
  const float* gat1_bias= (const float*)d_in[26];
  const float* skip1_w  = (const float*)d_in[27];
  const float* skip1_b  = (const float*)d_in[28];
  const float* bng1_g   = (const float*)d_in[29];
  const float* bng1_b   = (const float*)d_in[30];
  const float* gat2_w   = (const float*)d_in[31];
  const float* gat2_as  = (const float*)d_in[32];
  const float* gat2_ad  = (const float*)d_in[33];
  const float* gat2_bias= (const float*)d_in[34];
  const float* bng2_g   = (const float*)d_in[35];
  const float* bng2_b   = (const float*)d_in[36];
  const float* cls_w1   = (const float*)d_in[37];
  const float* cls_b1   = (const float*)d_in[38];
  const float* cls_w2   = (const float*)d_in[39];
  const float* cls_b2   = (const float*)d_in[40];
  float* out = (float*)d_out;

  // workspace layout (floats)
  float* ws = (float*)d_ws;
  float* yv      = ws;                    // 128*16
  float* feat    = yv      + B_ * 16;     // 2432*32
  float* scores  = feat    + NODES * 32;  // 2432
  float* h1      = scores  + NODES;       // 2432*512
  float* gat1out = h1      + NODES * HD_; // 2432*64
  float* skipm   = gat1out + NODES * D_;  // 2432*64
  float* hbuf    = skipm   + NODES * D_;  // 2432*64
  float* h2pre   = hbuf    + NODES * D_;  // 2432*512
  float* gat2out = h2pre   + NODES * HD_; // 2432*64
  float* hf      = gat2out + NODES * D_;  // 2432*64
  float* gmean   = hf      + NODES * D_;  // 128*64
  float* act     = gmean   + B_ * D_;     // 128*64

  // 1. frontend
  fe_kernel<<<B_, 256, 0, stream>>>(x, conv1_w, bn1_g, bn1_b, depth_w, bn2_g, bn2_b,
                                    se_w1, se_w2, sep_w, bn3_g, bn3_b, yv);
  // 2. multiscale
  ms_kernel<<<NODES, 256, 0, stream>>>(x, ms_w0, ms_g0, ms_b0, ms_w1, ms_g1, ms_b1,
                                       ms_w2, ms_g2, ms_b2, feat);
  // 3. feat assembly + edge scores
  feat_kernel<<<(NODES + 255) / 256, 256, 0, stream>>>(yv, adj_w, adj_b, feat, scores);
  // 4. h1 = feat @ gat1_w  [2432,32]x[32,512]: 152 M-tiles x 4 panel groups
  wmma_gemm<32><<<152 * 4, 256, 0, stream>>>(feat, gat1_w, h1, NODES, HD_);
  // 5. skip = feat @ skip1_w [2432,32]x[32,64]: 152 x 1
  wmma_gemm<32><<<152 * 1, 256, 0, stream>>>(feat, skip1_w, skipm, NODES, D_);
  // 6. GAT layer 1 attention
  gat_kernel<<<B_, 256, 0, stream>>>(h1, gat1_as, gat1_ad, scores, gat1_bias, gat1out);
  // 7. h = gelu(bn(gat1out) + skip + b)
  post1_kernel<<<(NODES * D_ + 255) / 256, 256, 0, stream>>>(gat1out, skipm, bng1_g, bng1_b,
                                                             skip1_b, hbuf);
  // 8. h2pre = h @ gat2_w [2432,64]x[64,512]: 152 x 4
  wmma_gemm<64><<<152 * 4, 256, 0, stream>>>(hbuf, gat2_w, h2pre, NODES, HD_);
  // 9. GAT layer 2 attention
  gat_kernel<<<B_, 256, 0, stream>>>(h2pre, gat2_as, gat2_ad, scores, gat2_bias, gat2out);
  // 10. h = gelu(bn(h2) + h2)
  post2_kernel<<<(NODES * D_ + 255) / 256, 256, 0, stream>>>(gat2out, bng2_g, bng2_b, hf);
  // 11. graph mean pool
  pool_kernel<<<(B_ * D_ + 255) / 256, 256, 0, stream>>>(hf, gmean);
  // 12. act = g @ cls_w1 [128,64]x[64,64]: 8 x 1
  wmma_gemm<64><<<8, 256, 0, stream>>>(gmean, cls_w1, act, B_, D_);
  // 13. out = gelu(act + b1) @ cls_w2 + b2
  cls_kernel<<<1, 256, 0, stream>>>(act, cls_b1, cls_w2, cls_b2, out);
}
